// PromptEnhancerDecoderSID_56255481643687
// MI455X (gfx1250) — matrix operand order repaired and checked by reading, not compiled, not run on hardware
//
#include <hip/hip_runtime.h>
#include <hip/hip_bf16.h>

// ---------------------------------------------------------------------------
// PromptEnhancerDecoderSID for MI455X (gfx1250, wave32, WMMA bf16)
// D=256 H=8 DH=32 DFF=2048 T=77 M=800 B=64, per-sample weights via sid.
// All GEMMs + attention run on v_wmma_f32_16x16x32_bf16 (f32->bf16 staged in
// double-buffered LDS; one barrier per K-step so HBM loads hide under WMMA).
// ---------------------------------------------------------------------------

#define D_    256
#define H_    8
#define DH_   32
#define DFF_  2048
#define T_    77
#define M1_   400
#define M2_   400
#define MM_   800
#define B_    64
#define NQT_  5            // ceil(77/16)
#define EPS_  1e-5f
#define NEG_  (-1.0e9f)

typedef __attribute__((ext_vector_type(16))) __bf16 bf16x16;
typedef __attribute__((ext_vector_type(8)))  __bf16 bf16x8;
typedef __attribute__((ext_vector_type(8)))  float  f32x8;

union BF16Frag { bf16x16 v; bf16x8 h[2]; };

static __device__ inline bf16x8 cvt8(float4 a, float4 b) {
  bf16x8 r;
  r[0] = (__bf16)a.x; r[1] = (__bf16)a.y; r[2] = (__bf16)a.z; r[3] = (__bf16)a.w;
  r[4] = (__bf16)b.x; r[5] = (__bf16)b.y; r[6] = (__bf16)b.z; r[7] = (__bf16)b.w;
  return r;
}

static __device__ inline bf16x16 ld_frag(const __bf16* lo, const __bf16* hi) {
  BF16Frag f;
  f.h[0] = *(const bf16x8*)lo;
  f.h[1] = *(const bf16x8*)hi;
  return f.v;
}

// ---------------------------------------------------------------------------
// Batched GEMM:  Y[b] = X[b] @ W[sid[b]]^T + bias[sid[b]]  (optional ReLU)
// X: rows x K (ld=K), W: NC x K (row-major, K contiguous), Y: rows x ldY
// Grid: (NC/64, ceil(rows/128), B). Block: 256 (8 waves), wave tile 32x32.
// Double-buffered LDS, one barrier per 32-wide K step.
// ---------------------------------------------------------------------------
__global__ __launch_bounds__(256) void gemm_bf16_kernel(
    const float* __restrict__ X, int xbs, int rows, int K,
    const float* __restrict__ W, int wstride,
    const float* __restrict__ bias, int bstride,
    const int* __restrict__ sid,
    float* __restrict__ Y, int ybs, int ldY, int relu)
{
  __shared__ __bf16 lA[2][128][32];
  __shared__ __bf16 lB[2][64][32];

  const int tid  = threadIdx.x;
  const int wv   = tid >> 5;
  const int lane = tid & 31;
  const int half = lane >> 4;
  const int l16  = lane & 15;
  const int wm   = wv & 3;       // 4 wave rows  -> 128 M
  const int wn   = wv >> 2;      // 2 wave cols  -> 64 N
  const int bb   = blockIdx.z;
  const int bM   = blockIdx.y * 128;
  const int bN   = blockIdx.x * 64;

  const int wid = sid[bb];
  const float* Xb    = X + (size_t)bb * xbs;
  const float* Wb    = W + (size_t)wid * wstride;
  const float* biasb = bias + (size_t)wid * bstride;

  const f32x8 z8 = {0.f,0.f,0.f,0.f,0.f,0.f,0.f,0.f};
  f32x8 acc[2][2] = {{z8, z8}, {z8, z8}};

  const int ar  = tid >> 1;             // 0..127
  const int ac  = (tid & 1) * 16;       // 0 or 16
  const int br  = tid >> 2;             // 0..63
  const int bc  = (tid & 3) * 8;        // 0,8,16,24
  const int arg = bM + ar;
  const int arc = (arg < rows) ? arg : (rows - 1);   // clamp: no exec juggling
  const float* aptr = Xb + (size_t)arc * K + ac;
  const float* bptr = Wb + (size_t)(bN + br) * K + bc;

  float4 a0, a1, a2, a3, b0, b1;
#define LOAD_TILE(KT)                                              \
  do {                                                             \
    const float* pa_ = aptr + (KT);                                \
    a0 = *(const float4*)(pa_ + 0);                                \
    a1 = *(const float4*)(pa_ + 4);                                \
    a2 = *(const float4*)(pa_ + 8);                                \
    a3 = *(const float4*)(pa_ + 12);                               \
    const float* pb_ = bptr + (KT);                                \
    b0 = *(const float4*)(pb_ + 0);                                \
    b1 = *(const float4*)(pb_ + 4);                                \
  } while (0)

  LOAD_TILE(0);          // prologue
  int p = 0;

  for (int kt = 0; kt < K; kt += 32) {
    // commit staged registers of tile kt into LDS buffer p
    *(bf16x8*)&lA[p][ar][ac]     = cvt8(a0, a1);
    *(bf16x8*)&lA[p][ar][ac + 8] = cvt8(a2, a3);
    *(bf16x8*)&lB[p][br][bc]     = cvt8(b0, b1);
    __syncthreads();

    // start global loads for tile kt+32 (hidden under the WMMAs below)
    if (kt + 32 < K) {
      LOAD_TILE(kt + 32);
      __builtin_prefetch(aptr + kt + 64, 0, 3);
      __builtin_prefetch(bptr + kt + 64, 0, 3);
    }

    bf16x16 af[2], bfr[2];
#pragma unroll
    for (int i = 0; i < 2; ++i) {
      // 16-bit A 16x32 layout: lane row = l16; K chunks half*8 and 16+half*8
      const __bf16* base = &lA[p][wm * 32 + i * 16 + l16][half * 8];
      af[i] = ld_frag(base, base + 16);
    }
#pragma unroll
    for (int j = 0; j < 2; ++j) {
      // B 32x16 layout: lane col = l16; K chunks half*8 and 16+half*8
      const __bf16* base = &lB[p][wn * 32 + j * 16 + l16][half * 8];
      bfr[j] = ld_frag(base, base + 16);
    }
#pragma unroll
    for (int i = 0; i < 2; ++i)
#pragma unroll
      for (int j = 0; j < 2; ++j)
        acc[i][j] = __builtin_amdgcn_wmma_f32_16x16x32_bf16(
            false, af[i], false, bfr[j], (short)0, acc[i][j], false, false);
    p ^= 1;
  }
#undef LOAD_TILE

  // Epilogue: C layout row = half*8 + vr, col = l16
#pragma unroll
  for (int i = 0; i < 2; ++i) {
#pragma unroll
    for (int j = 0; j < 2; ++j) {
      const int cg = bN + wn * 32 + j * 16 + l16;
      const float bv = biasb[cg];
#pragma unroll
      for (int vr = 0; vr < 8; ++vr) {
        const int rg = bM + wm * 32 + i * 16 + half * 8 + vr;
        if (rg < rows) {
          float val = acc[i][j][vr] + bv;
          if (relu) val = fmaxf(val, 0.f);
          Y[(size_t)bb * ybs + (size_t)rg * ldY + cg] = val;
        }
      }
    }
  }
}

// ---------------------------------------------------------------------------
// Flash attention per (b,h): softmax(Q K^T / sqrt(DH) + mask) V
// Grid (H, B), block 128 (4 waves). 16-query tiles per wave, 32-key blocks.
// ---------------------------------------------------------------------------
__global__ __launch_bounds__(128) void attn_kernel(
    const float* __restrict__ Q, int qbs, int ldq,
    const float* __restrict__ Kp, int kbs, int ldk,
    const float* __restrict__ Vp, int vbs, int ldv,
    const unsigned char* __restrict__ mask, int mbs,
    int Lk, float* __restrict__ Out, int obs)
{
  __shared__ __bf16 lQ[80][32];        // queries, pre-scaled by 1/sqrt(32)
  __shared__ __bf16 lK[32][32];        // [key][dh]
  __shared__ __bf16 lVt[32][32];       // [dh][key] (transposed)
  __shared__ __bf16 lP[4][16][32];     // per-wave P scratch
  __shared__ float  lmadd[32];         // additive mask per key of block

  const int h  = blockIdx.x;
  const int bb = blockIdx.y;
  const int tid  = threadIdx.x;
  const int wv   = tid >> 5;
  const int lane = tid & 31;
  const int half = lane >> 4;
  const int l16  = lane & 15;

  const float* q = Q  + (size_t)bb * qbs + h * DH_;
  const float* k = Kp + (size_t)bb * kbs + h * DH_;
  const float* v = Vp + (size_t)bb * vbs + h * DH_;
  const unsigned char* msk = mask + (size_t)bb * mbs;

  for (int idx = tid; idx < 80 * 32; idx += 128) {
    const int r = idx >> 5, c = idx & 31;
    const float val = (r < T_) ? q[(size_t)r * ldq + c] * 0.17677669529663687f : 0.f;
    lQ[r][c] = (__bf16)val;
  }
  __syncthreads();

  const f32x8 z8 = {0.f,0.f,0.f,0.f,0.f,0.f,0.f,0.f};
  bf16x16 qf[2];
  f32x8 accO[2][2];
  float mrow[2][8], lrow[2][8];
  bool act[2];
#pragma unroll
  for (int qi = 0; qi < 2; ++qi) {
    const int qt0 = wv + qi * 4;
    act[qi] = (qt0 < NQT_);
    const int qt = act[qi] ? qt0 : 0;          // clamp: keep EXEC full for WMMA
    const __bf16* base = &lQ[qt * 16 + l16][half * 8];
    qf[qi] = ld_frag(base, base + 16);
    accO[qi][0] = z8; accO[qi][1] = z8;
#pragma unroll
    for (int vr = 0; vr < 8; ++vr) { mrow[qi][vr] = -1e30f; lrow[qi][vr] = 0.f; }
  }

  const int nkb = (Lk + 31) >> 5;
  const int kr = tid >> 2;            // 0..31 key within block
  const int kc = (tid & 3) * 8;       // dh chunk

  for (int kb = 0; kb < nkb; ++kb) {
    {   // stage K, V (transposed), mask; clamped loads, masking handles padding
      const int key  = kb * 32 + kr;
      const int keyc = (key < Lk) ? key : (Lk - 1);
      const float* kp = k + (size_t)keyc * ldk + kc;
      float4 k0 = *(const float4*)(kp + 0);
      float4 k1 = *(const float4*)(kp + 4);
      const float* vp = v + (size_t)keyc * ldv + kc;
      float4 v0 = *(const float4*)(vp + 0);
      float4 v1 = *(const float4*)(vp + 4);
      *(bf16x8*)&lK[kr][kc] = cvt8(k0, k1);
      lVt[kc + 0][kr] = (__bf16)v0.x; lVt[kc + 1][kr] = (__bf16)v0.y;
      lVt[kc + 2][kr] = (__bf16)v0.z; lVt[kc + 3][kr] = (__bf16)v0.w;
      lVt[kc + 4][kr] = (__bf16)v1.x; lVt[kc + 5][kr] = (__bf16)v1.y;
      lVt[kc + 6][kr] = (__bf16)v1.z; lVt[kc + 7][kr] = (__bf16)v1.w;
      if (tid < 32) {
        const int key2 = kb * 32 + tid;
        lmadd[tid] = (key2 < Lk && msk[key2] == 0) ? 0.f : NEG_;
      }
    }
    __syncthreads();

#pragma unroll
    for (int qi = 0; qi < 2; ++qi) {
      // S = Q K^T for 32 keys (two 16-key wmma)
      f32x8 s0 = z8, s1 = z8;
      const __bf16* kb0 = &lK[l16][half * 8];
      const __bf16* kb1 = &lK[16 + l16][half * 8];
      bf16x16 kf0 = ld_frag(kb0, kb0 + 16);
      bf16x16 kf1 = ld_frag(kb1, kb1 + 16);
      s0 = __builtin_amdgcn_wmma_f32_16x16x32_bf16(false, qf[qi], false, kf0, (short)0, s0, false, false);
      s1 = __builtin_amdgcn_wmma_f32_16x16x32_bf16(false, qf[qi], false, kf1, (short)0, s1, false, false);

      const float madd0 = lmadd[l16];
      const float madd1 = lmadd[16 + l16];
      float p0[8], p1[8], sc[8];
#pragma unroll
      for (int vr = 0; vr < 8; ++vr) {
        const float sa = s0[vr] + madd0;
        const float sb = s1[vr] + madd1;
        float mx = fmaxf(sa, sb);
        mx = fmaxf(mx, __shfl_xor(mx, 1, 32));
        mx = fmaxf(mx, __shfl_xor(mx, 2, 32));
        mx = fmaxf(mx, __shfl_xor(mx, 4, 32));
        mx = fmaxf(mx, __shfl_xor(mx, 8, 32));
        const float nm = fmaxf(mrow[qi][vr], mx);
        const float scale = __expf(mrow[qi][vr] - nm);
        mrow[qi][vr] = nm;
        const float e0 = __expf(sa - nm);
        const float e1 = __expf(sb - nm);
        float sum = e0 + e1;
        sum += __shfl_xor(sum, 1, 32);
        sum += __shfl_xor(sum, 2, 32);
        sum += __shfl_xor(sum, 4, 32);
        sum += __shfl_xor(sum, 8, 32);
        lrow[qi][vr] = lrow[qi][vr] * scale + sum;
        sc[vr] = scale; p0[vr] = e0; p1[vr] = e1;
      }
#pragma unroll
      for (int vr = 0; vr < 8; ++vr) {
        accO[qi][0][vr] *= sc[vr];
        accO[qi][1][vr] *= sc[vr];
        lP[wv][half * 8 + vr][l16]      = (__bf16)p0[vr];   // C-layout -> LDS
        lP[wv][half * 8 + vr][16 + l16] = (__bf16)p1[vr];
      }
      asm volatile("s_wait_dscnt 0" ::: "memory");          // wave-local P round-trip
      const __bf16* pb = &lP[wv][l16][half * 8];            // reload as A-layout
      bf16x16 pf = ld_frag(pb, pb + 16);
      const __bf16* vb0 = &lVt[l16][half * 8];
      const __bf16* vb1 = &lVt[16 + l16][half * 8];
      bf16x16 vf0 = ld_frag(vb0, vb0 + 16);
      bf16x16 vf1 = ld_frag(vb1, vb1 + 16);
      accO[qi][0] = __builtin_amdgcn_wmma_f32_16x16x32_bf16(false, pf, false, vf0, (short)0, accO[qi][0], false, false);
      accO[qi][1] = __builtin_amdgcn_wmma_f32_16x16x32_bf16(false, pf, false, vf1, (short)0, accO[qi][1], false, false);
    }
    __syncthreads();
  }

#pragma unroll
  for (int qi = 0; qi < 2; ++qi) {
    const int qt0 = wv + qi * 4;
    if (!act[qi]) continue;
#pragma unroll
    for (int jj = 0; jj < 2; ++jj)
#pragma unroll
      for (int vr = 0; vr < 8; ++vr) {
        const int qrow = qt0 * 16 + half * 8 + vr;
        if (qrow < T_) {
          Out[(size_t)bb * obs + (size_t)qrow * D_ + h * DH_ + jj * 16 + l16] =
              accO[qi][jj][vr] / lrow[qi][vr];
        }
      }
  }
}

// ---------------------------------------------------------------------------
// Y = LayerNorm(A + R) * w[sid] + b[sid]   (one wave per 256-wide row)
// ---------------------------------------------------------------------------
__global__ __launch_bounds__(256) void add_ln_kernel(
    const float* __restrict__ A, const float* __restrict__ R,
    const float* __restrict__ Wl, const float* __restrict__ Bl,
    const int* __restrict__ sid, float* __restrict__ Y, int rowsTotal)
{
  const int wv = threadIdx.x >> 5, lane = threadIdx.x & 31;
  const int row = blockIdx.x * 8 + wv;
  if (row >= rowsTotal) return;
  const int bb = row / T_;
  const int wid = sid[bb];
  const float* wp = Wl + (size_t)wid * D_;
  const float* bp = Bl + (size_t)wid * D_;
  const size_t base = (size_t)row * D_ + lane * 8;

  float4 x0 = *(const float4*)(A + base);
  float4 x1 = *(const float4*)(A + base + 4);
  float4 r0 = *(const float4*)(R + base);
  float4 r1 = *(const float4*)(R + base + 4);
  float vbuf[8] = {x0.x + r0.x, x0.y + r0.y, x0.z + r0.z, x0.w + r0.w,
                   x1.x + r1.x, x1.y + r1.y, x1.z + r1.z, x1.w + r1.w};
  float s = 0.f;
#pragma unroll
  for (int i = 0; i < 8; ++i) s += vbuf[i];
  s += __shfl_xor(s, 1, 32); s += __shfl_xor(s, 2, 32);
  s += __shfl_xor(s, 4, 32); s += __shfl_xor(s, 8, 32); s += __shfl_xor(s, 16, 32);
  const float mean = s * (1.f / D_);
  float sv = 0.f;
#pragma unroll
  for (int i = 0; i < 8; ++i) { const float d = vbuf[i] - mean; sv += d * d; }
  sv += __shfl_xor(sv, 1, 32); sv += __shfl_xor(sv, 2, 32);
  sv += __shfl_xor(sv, 4, 32); sv += __shfl_xor(sv, 8, 32); sv += __shfl_xor(sv, 16, 32);
  const float rstd = rsqrtf(sv * (1.f / D_) + EPS_);

  float4 w0 = *(const float4*)(wp + lane * 8);
  float4 w1 = *(const float4*)(wp + lane * 8 + 4);
  float4 b0 = *(const float4*)(bp + lane * 8);
  float4 b1 = *(const float4*)(bp + lane * 8 + 4);
  float4 o0, o1;
  o0.x = (vbuf[0] - mean) * rstd * w0.x + b0.x;
  o0.y = (vbuf[1] - mean) * rstd * w0.y + b0.y;
  o0.z = (vbuf[2] - mean) * rstd * w0.z + b0.z;
  o0.w = (vbuf[3] - mean) * rstd * w0.w + b0.w;
  o1.x = (vbuf[4] - mean) * rstd * w1.x + b1.x;
  o1.y = (vbuf[5] - mean) * rstd * w1.y + b1.y;
  o1.z = (vbuf[6] - mean) * rstd * w1.z + b1.z;
  o1.w = (vbuf[7] - mean) * rstd * w1.w + b1.w;
  *(float4*)(Y + base) = o0;
  *(float4*)(Y + base + 4) = o1;
}

// ---------------------------------------------------------------------------
// Concat helpers: mem = [im1 ; im2], mmask = [m1 ; m2]
// ---------------------------------------------------------------------------
__global__ void concat_mem_kernel(const float4* __restrict__ a,
                                  const float4* __restrict__ c,
                                  float4* __restrict__ mem)
{
  const int i = blockIdx.x * 256 + threadIdx.x;
  const int total = B_ * MM_ * (D_ / 4);
  if (i >= total) return;
  const int rowv = i / (D_ / 4);
  const int cc = i - rowv * (D_ / 4);
  const int bb = rowv / MM_;
  const int r = rowv - bb * MM_;
  mem[i] = (r < M1_) ? a[((size_t)bb * M1_ + r) * (D_ / 4) + cc]
                     : c[((size_t)bb * M2_ + (r - M1_)) * (D_ / 4) + cc];
}

__global__ void concat_mask_kernel(const unsigned char* __restrict__ a,
                                   const unsigned char* __restrict__ c,
                                   unsigned char* __restrict__ m)
{
  const int i = blockIdx.x * 256 + threadIdx.x;
  if (i >= B_ * MM_) return;
  const int bb = i / MM_;
  const int r = i - bb * MM_;
  m[i] = (r < M1_) ? a[bb * M1_ + r] : c[bb * M2_ + (r - M1_)];
}

// ---------------------------------------------------------------------------
extern "C" void kernel_launch(void* const* d_in, const int* in_sizes, int n_in,
                              void* d_out, int out_size, void* d_ws, size_t ws_size,
                              hipStream_t stream) {
  (void)in_sizes; (void)n_in; (void)out_size; (void)ws_size;

  const float* text          = (const float*)d_in[0];
  const unsigned char* tmask = (const unsigned char*)d_in[1];
  const float* im1           = (const float*)d_in[2];
  const unsigned char* im1m  = (const unsigned char*)d_in[3];
  const float* im2           = (const float*)d_in[4];
  const unsigned char* im2m  = (const unsigned char*)d_in[5];
  const int* sid             = (const int*)d_in[6];
  const float* sa_in_w  = (const float*)d_in[7];
  const float* sa_in_b  = (const float*)d_in[8];
  const float* sa_out_w = (const float*)d_in[9];
  const float* sa_out_b = (const float*)d_in[10];
  const float* ca_in_w  = (const float*)d_in[11];
  const float* ca_in_b  = (const float*)d_in[12];
  const float* ca_out_w = (const float*)d_in[13];
  const float* ca_out_b = (const float*)d_in[14];
  const float* l1w = (const float*)d_in[15];
  const float* l1b = (const float*)d_in[16];
  const float* l2w = (const float*)d_in[17];
  const float* l2b = (const float*)d_in[18];
  const float* n1w = (const float*)d_in[19];
  const float* n1b = (const float*)d_in[20];
  const float* n2w = (const float*)d_in[21];
  const float* n2b = (const float*)d_in[22];
  const float* n3w = (const float*)d_in[23];
  const float* n3b = (const float*)d_in[24];
  float* out = (float*)d_out;

  // Workspace carve (~240 MB)
  char* p = (char*)d_ws;
  auto carve = [&](size_t bytes) { char* r = p; p += (bytes + 255) & ~(size_t)255; return r; };
  float* mem            = (float*)carve((size_t)B_ * MM_ * D_ * 4);
  unsigned char* mmask  = (unsigned char*)carve((size_t)B_ * MM_);
  float* qkv            = (float*)carve((size_t)B_ * T_ * 3 * D_ * 4);
  float* attnb          = (float*)carve((size_t)B_ * T_ * D_ * 4);
  float* tmp            = (float*)carve((size_t)B_ * T_ * D_ * 4);
  float* x1             = (float*)carve((size_t)B_ * T_ * D_ * 4);
  float* qca            = (float*)carve((size_t)B_ * T_ * D_ * 4);
  float* kvca           = (float*)carve((size_t)B_ * MM_ * 2 * D_ * 4);
  float* x2             = (float*)carve((size_t)B_ * T_ * D_ * 4);
  float* ff1            = (float*)carve((size_t)B_ * T_ * DFF_ * 4);

  // 0) concat memory tokens + masks
  {
    const int total = B_ * MM_ * (D_ / 4);
    concat_mem_kernel<<<(total + 255) / 256, 256, 0, stream>>>(
        (const float4*)im1, (const float4*)im2, (float4*)mem);
    const int tm = B_ * MM_;
    concat_mask_kernel<<<(tm + 255) / 256, 256, 0, stream>>>(im1m, im2m, mmask);
  }

  // 1) SA QKV projection: (77x256) @ (768x256)^T -> 77x768
  gemm_bf16_kernel<<<dim3(768 / 64, 1, B_), 256, 0, stream>>>(
      text, T_ * D_, T_, D_, sa_in_w, 3 * D_ * D_, sa_in_b, 3 * D_, sid,
      qkv, T_ * 3 * D_, 3 * D_, 0);

  // 2) SA attention (Lk = 77)
  attn_kernel<<<dim3(H_, B_), 128, 0, stream>>>(
      qkv, T_ * 3 * D_, 3 * D_, qkv + D_, T_ * 3 * D_, 3 * D_,
      qkv + 2 * D_, T_ * 3 * D_, 3 * D_, tmask, T_, T_, attnb, T_ * D_);

  // 3) SA out projection
  gemm_bf16_kernel<<<dim3(D_ / 64, 1, B_), 256, 0, stream>>>(
      attnb, T_ * D_, T_, D_, sa_out_w, D_ * D_, sa_out_b, D_, sid,
      tmp, T_ * D_, D_, 0);

  // 4) x1 = LN1(text + SA)
  add_ln_kernel<<<(B_ * T_ + 7) / 8, 256, 0, stream>>>(text, tmp, n1w, n1b, sid, x1, B_ * T_);

  // 5) CA Q projection (first 256 rows of ca_in_w)
  gemm_bf16_kernel<<<dim3(D_ / 64, 1, B_), 256, 0, stream>>>(
      x1, T_ * D_, T_, D_, ca_in_w, 3 * D_ * D_, ca_in_b, 3 * D_, sid,
      qca, T_ * D_, D_, 0);

  // 6) CA KV projection: (800x256) @ (512x256)^T -> 800x512
  gemm_bf16_kernel<<<dim3(512 / 64, (MM_ + 127) / 128, B_), 256, 0, stream>>>(
      mem, MM_ * D_, MM_, D_, ca_in_w + D_ * D_, 3 * D_ * D_, ca_in_b + D_, 3 * D_, sid,
      kvca, MM_ * 2 * D_, 2 * D_, 0);

  // 7) CA attention (Lk = 800)
  attn_kernel<<<dim3(H_, B_), 128, 0, stream>>>(
      qca, T_ * D_, D_, kvca, MM_ * 2 * D_, 2 * D_,
      kvca + D_, MM_ * 2 * D_, 2 * D_, mmask, MM_, MM_, attnb, T_ * D_);

  // 8) CA out projection
  gemm_bf16_kernel<<<dim3(D_ / 64, 1, B_), 256, 0, stream>>>(
      attnb, T_ * D_, T_, D_, ca_out_w, D_ * D_, ca_out_b, D_, sid,
      tmp, T_ * D_, D_, 0);

  // 9) x2 = LN2(x1 + CA)
  add_ln_kernel<<<(B_ * T_ + 7) / 8, 256, 0, stream>>>(x1, tmp, n2w, n2b, sid, x2, B_ * T_);

  // 10) FFN1 + ReLU: (77x256) @ (2048x256)^T
  gemm_bf16_kernel<<<dim3(DFF_ / 64, 1, B_), 256, 0, stream>>>(
      x2, T_ * D_, T_, D_, l1w, DFF_ * D_, l1b, DFF_, sid,
      ff1, T_ * DFF_, DFF_, 1);

  // 11) FFN2: (77x2048) @ (256x2048)^T
  gemm_bf16_kernel<<<dim3(D_ / 64, 1, B_), 256, 0, stream>>>(
      ff1, T_ * DFF_, T_, DFF_, l2w, D_ * DFF_, l2b, D_, sid,
      tmp, T_ * D_, D_, 0);

  // 12) out = LN3(x2 + ff)
  add_ln_kernel<<<(B_ * T_ + 7) / 8, 256, 0, stream>>>(x2, tmp, n3w, n3b, sid, out, B_ * T_);
}
